// Attention_74354473828561
// MI455X (gfx1250) — matrix operand order repaired and checked by reading, not compiled, not run on hardware
//
#include <hip/hip_runtime.h>
#include <hip/hip_bf16.h>
#include <cstdint>

typedef __bf16 bf16;
typedef __attribute__((ext_vector_type(16))) __bf16 v16bf;
typedef __attribute__((ext_vector_type(8)))  __bf16 v8bf;
typedef __attribute__((ext_vector_type(8)))  float  v8f;
typedef __attribute__((ext_vector_type(4)))  unsigned int ui4;
typedef __attribute__((ext_vector_type(8)))  int i8x;
typedef __attribute__((ext_vector_type(4)))  int i4x;

// ---------------- problem constants ----------------
#define BB   2
#define SS   1024
#define DD   768
#define MM   4
#define HH   12
#define HD   64
#define LDQ  2304          // 3*D, qkv row stride
#define SCALE 0.125f       // hd^-0.5

#define HAVE_TDM __has_builtin(__builtin_amdgcn_tensor_load_to_lds)

// ---------------- WMMA helpers ----------------
__device__ __forceinline__ v8f wmma_bf16(v16bf a, v16bf b, v8f c) {
  return __builtin_amdgcn_wmma_f32_16x16x32_bf16(false, a, false, b, (short)0, c, false, false);
}

// A-operand (16x32, M x K): lane<16: row=lane, K 0..7 & 16..23 ; lane>=16: row=lane-16, K 8..15 & 24..31
__device__ __forceinline__ v16bf load_aop(const bf16* base, int ld) {
  const int lane = threadIdx.x & 31;
  const int row  = lane & 15;
  const int kb   = (lane >> 4) * 8;
  const bf16* p  = base + (size_t)row * ld + kb;
  v8bf lo = *(const v8bf*)p;
  v8bf hi = *(const v8bf*)(p + 16);
  v16bf r;
#pragma unroll
  for (int i = 0; i < 8; ++i) { r[i] = lo[i]; r[8 + i] = hi[i]; }
  return r;
}

// B-operand (32x16, K x N), source stores B^T rows contiguously (row n = column n of B):
// lane<16: col=lane, K 0..7 & 16..23 ; lane>=16: col=lane-16, K 8..15 & 24..31
__device__ __forceinline__ v16bf load_bop(const bf16* base, int ld) {
  const int lane = threadIdx.x & 31;
  const int col  = lane & 15;
  const int kb   = (lane >> 4) * 8;
  const bf16* p  = base + (size_t)col * ld + kb;
  v8bf lo = *(const v8bf*)p;
  v8bf hi = *(const v8bf*)(p + 16);
  v16bf r;
#pragma unroll
  for (int i = 0; i < 8; ++i) { r[i] = lo[i]; r[8 + i] = hi[i]; }
  return r;
}

#if HAVE_TDM
// TDM 2D tile load: tile_d0 (contiguous elems, bf16) x tile_d1 rows, memory row
// stride = row_stride elems, into LDS at lds_byte_off. LDS padding: 4 DWORDs after
// every 16 DWORDs stored -> effective LDS row stride = 40 bf16 elements.
__device__ __forceinline__ void tdm_load_tile_2d(unsigned lds_byte_off, const void* g,
                                                 unsigned tile_d0, unsigned tile_d1,
                                                 unsigned long long row_stride) {
  const unsigned long long ga = (unsigned long long)(uintptr_t)g;
  ui4 g0;
  g0[0] = 1u;                                              // count=1, user descriptor
  g0[1] = lds_byte_off;                                    // lds_addr
  g0[2] = (unsigned)ga;                                    // global_addr[31:0]
  g0[3] = (unsigned)((ga >> 32) & 0x01ffffffu) | (2u << 30); // addr[56:32] | type=2
  i8x g1;
  g1[0] = (int)((1u << 16)     // data_size = 1 -> 2 bytes
              | (1u << 20)     // pad_enable
              | (3u << 22)     // pad_interval = 16 DWORDs
              | (3u << 25));   // pad_amount   = 4 DWORDs
  g1[1] = (int)((tile_d0 & 0xffffu) << 16);                              // tensor_dim0 lo
  g1[2] = (int)(((tile_d0 >> 16) & 0xffffu) | ((tile_d1 & 0xffffu) << 16)); // dim0 hi | dim1 lo
  g1[3] = (int)(((tile_d1 >> 16) & 0xffffu) | ((tile_d0 & 0xffffu) << 16)); // dim1 hi | tile_dim0
  g1[4] = (int)(tile_d1 & 0xffffu);                                      // tile_dim1 ; tile_dim2=0
  g1[5] = (int)(row_stride & 0xffffffffull);                             // dim0_stride lo
  g1[6] = (int)((row_stride >> 32) & 0xffffull);                         // dim0_stride hi | dim1_stride lo
  g1[7] = 0;
  i4x z = {};
#if defined(__clang_major__) && (__clang_major__ >= 23)
  i8x z8 = {};
  __builtin_amdgcn_tensor_load_to_lds(g0, g1, z, z, z8, 0);
#else
  __builtin_amdgcn_tensor_load_to_lds(g0, g1, z, z, 0);
#endif
}
#endif

// ---------------- fp32 -> bf16 convert ----------------
__global__ void k_cvt(const float* __restrict__ s, bf16* __restrict__ d, int n) {
  int i = blockIdx.x * blockDim.x + threadIdx.x;
  int stride = gridDim.x * blockDim.x;
  for (; i < n; i += stride) d[i] = (bf16)s[i];
}

// ---------------- generic GEMM: C[M,N] = A[M,K] * W[N,K]^T (+bias) ----------------
// A, W bf16 row-major (K-major). 128x128 block, 8 waves (2M x 4N), each wave 64x32.
// LDS tiles are staged by the Tensor Data Mover (double buffered); waves only compute.
template <bool OUTF32>
__global__ __launch_bounds__(256)
void k_gemm_nt(const bf16* __restrict__ A, const bf16* __restrict__ W,
               bf16* __restrict__ Cb, float* __restrict__ Cf,
               const float* __restrict__ bias, int N, int K) {
  __shared__ __align__(16) bf16 As[2][128 * 40];
  __shared__ __align__(16) bf16 Bs[2][128 * 40];
  const int tid  = threadIdx.x;
  const int lane = tid & 31;
  const int w    = tid >> 5;
  const int wm   = w & 1;
  const int wn   = w >> 1;
  const int m0   = blockIdx.y * 128;
  const int n0   = blockIdx.x * 128;

  v8f acc[4][2] = {};
  const int nk = K / 32;

#if HAVE_TDM
  const unsigned asoff[2] = { (unsigned)(uintptr_t)&As[0][0], (unsigned)(uintptr_t)&As[1][0] };
  const unsigned bsoff[2] = { (unsigned)(uintptr_t)&Bs[0][0], (unsigned)(uintptr_t)&Bs[1][0] };
  if (w == 0) {
    tdm_load_tile_2d(asoff[0], A + (size_t)m0 * K, 32, 128, (unsigned long long)K);
    tdm_load_tile_2d(bsoff[0], W + (size_t)n0 * K, 32, 128, (unsigned long long)K);
  }
  for (int it = 0; it < nk; ++it) {
    const int cur = it & 1;
    __syncthreads();                 // prior reads of the next-issue buffer are done
    if (w == 0) {
      if (it + 1 < nk) {
        const size_t k1 = (size_t)(it + 1) * 32;
        tdm_load_tile_2d(asoff[cur ^ 1], A + (size_t)m0 * K + k1, 32, 128, (unsigned long long)K);
        tdm_load_tile_2d(bsoff[cur ^ 1], W + (size_t)n0 * K + k1, 32, 128, (unsigned long long)K);
        __builtin_amdgcn_s_wait_tensorcnt(2);   // current buffer's pair has landed
      } else {
        __builtin_amdgcn_s_wait_tensorcnt(0);
      }
    }
    __syncthreads();                 // broadcast completion to all waves

    v16bf av[4], bv[2];
#pragma unroll
    for (int i = 0; i < 4; ++i) av[i] = load_aop(&As[cur][(wm * 64 + i * 16) * 40], 40);
#pragma unroll
    for (int j = 0; j < 2; ++j) bv[j] = load_bop(&Bs[cur][(wn * 32 + j * 16) * 40], 40);
#pragma unroll
    for (int i = 0; i < 4; ++i)
#pragma unroll
      for (int j = 0; j < 2; ++j) acc[i][j] = wmma_bf16(av[i], bv[j], acc[i][j]);
  }
#else
  const int lr = tid >> 1;
  const int lc = (tid & 1) * 16;
  for (int k0 = 0; k0 < K; k0 += 32) {
    const bf16* ga = A + (size_t)(m0 + lr) * K + k0 + lc;
    const bf16* gb = W + (size_t)(n0 + lr) * K + k0 + lc;
    v8bf a0 = *(const v8bf*)ga, a1 = *(const v8bf*)(ga + 8);
    v8bf b0 = *(const v8bf*)gb, b1 = *(const v8bf*)(gb + 8);
    if (k0 + 32 < K) __builtin_prefetch(ga + 32, 0, 1);
    __syncthreads();
    *(v8bf*)&As[0][lr * 40 + lc] = a0; *(v8bf*)&As[0][lr * 40 + lc + 8] = a1;
    *(v8bf*)&Bs[0][lr * 40 + lc] = b0; *(v8bf*)&Bs[0][lr * 40 + lc + 8] = b1;
    __syncthreads();
    v16bf av[4], bv[2];
#pragma unroll
    for (int i = 0; i < 4; ++i) av[i] = load_aop(&As[0][(wm * 64 + i * 16) * 40], 40);
#pragma unroll
    for (int j = 0; j < 2; ++j) bv[j] = load_bop(&Bs[0][(wn * 32 + j * 16) * 40], 40);
#pragma unroll
    for (int i = 0; i < 4; ++i)
#pragma unroll
      for (int j = 0; j < 2; ++j) acc[i][j] = wmma_bf16(av[i], bv[j], acc[i][j]);
  }
#endif

  const int cr = (lane >> 4) * 8;
  const int cc = lane & 15;
#pragma unroll
  for (int i = 0; i < 4; ++i)
#pragma unroll
    for (int j = 0; j < 2; ++j) {
      const int rbase = m0 + wm * 64 + i * 16 + cr;
      const int cbase = n0 + wn * 32 + j * 16 + cc;
#pragma unroll
      for (int v = 0; v < 8; ++v) {
        float val = acc[i][j][v];
        if constexpr (OUTF32) {
          if (bias) val += bias[cbase];
          Cf[(size_t)(rbase + v) * N + cbase] = val;
        } else {
          Cb[(size_t)(rbase + v) * N + cbase] = (bf16)val;
        }
      }
    }
}

// ---------------- head-extract + transpose: out[slab][h][c][s] = in[slab][s][h*64+c] ----------------
__global__ __launch_bounds__(256)
void k_transpose64(const bf16* __restrict__ in, bf16* __restrict__ out,
                   size_t in_slab, size_t out_slab, int ldin, int Slen) {
  __shared__ __align__(16) bf16 tile[64][72];
  const int s0 = blockIdx.x * 64;
  const int h  = blockIdx.y;
  const int z  = blockIdx.z;
  const bf16* src = in + (size_t)z * in_slab + h * 64;
  bf16* dst = out + (size_t)z * out_slab + (size_t)h * 64 * Slen;
  const int t = threadIdx.x;
  {
    const int r  = t >> 2;
    const int c0 = (t & 3) * 16;
    const bf16* p = src + (size_t)(s0 + r) * ldin + c0;
    *(v8bf*)&tile[r][c0]     = *(const v8bf*)p;
    *(v8bf*)&tile[r][c0 + 8] = *(const v8bf*)(p + 8);
  }
  __syncthreads();
  {
    const int c  = t >> 2;
    const int sc = (t & 3) * 16;
    v8bf o0, o1;
#pragma unroll
    for (int i = 0; i < 8; ++i) o0[i] = tile[sc + i][c];
#pragma unroll
    for (int i = 0; i < 8; ++i) o1[i] = tile[sc + 8 + i][c];
    bf16* q = dst + (size_t)c * Slen + s0 + sc;
    *(v8bf*)q       = o0;
    *(v8bf*)(q + 8) = o1;
  }
}

// ---------------- fused attention JVP ----------------
// grid (S/16, H, B), block 128 = 4 waves, wave index == tangent index m.
__global__ __launch_bounds__(128)
void k_attn_jvp(const bf16* __restrict__ qkv, const bf16* __restrict__ qkvt,
                const bf16* __restrict__ vT, const bf16* __restrict__ vtT,
                bf16* __restrict__ ao, bf16* __restrict__ aot) {
  const int b  = blockIdx.z;
  const int h  = blockIdx.y;
  const int q0 = blockIdx.x * 16;
  const int m    = threadIdx.x >> 5;
  const int lane = threadIdx.x & 31;

  __shared__ __align__(16) bf16 ldsP[4][16 * 32];
  __shared__ __align__(16) bf16 ldsR[4][16 * 32];

  const bf16* Qp  = qkv  + ((size_t)b * SS + q0) * LDQ + h * HD;
  const bf16* Kp  = qkv  + (size_t)b * SS * LDQ + DD + h * HD;
  const bf16* Qtp = qkvt + (((size_t)b * MM + m) * SS + q0) * LDQ + h * HD;
  const bf16* Ktp = qkvt + ((size_t)b * MM + m) * SS * LDQ + DD + h * HD;
  const bf16* Vp  = vT   + ((size_t)(b * HH + h)) * HD * SS;
  const bf16* Vtp = vtT  + (((size_t)b * MM + m) * HH + h) * (size_t)HD * SS;

  const v16bf qa0 = load_aop(Qp, LDQ),      qa1 = load_aop(Qp + 32, LDQ);
  const v16bf qt0 = load_aop(Qtp, LDQ),     qt1 = load_aop(Qtp + 32, LDQ);

  // ---- pass 1: row max of raw scores ----
  float mx[8];
#pragma unroll
  for (int v = 0; v < 8; ++v) mx[v] = -3.0e38f;
  for (int n0 = 0; n0 < SS; n0 += 16) {
    v16bf kb0 = load_bop(Kp + (size_t)n0 * LDQ, LDQ);
    v16bf kb1 = load_bop(Kp + (size_t)n0 * LDQ + 32, LDQ);
    v8f s = {};
    s = wmma_bf16(qa0, kb0, s);
    s = wmma_bf16(qa1, kb1, s);
#pragma unroll
    for (int v = 0; v < 8; ++v) mx[v] = fmaxf(mx[v], s[v]);
  }
#pragma unroll
  for (int v = 0; v < 8; ++v) {
    mx[v] = fmaxf(mx[v], __shfl_xor(mx[v], 1, 16));
    mx[v] = fmaxf(mx[v], __shfl_xor(mx[v], 2, 16));
    mx[v] = fmaxf(mx[v], __shfl_xor(mx[v], 4, 16));
    mx[v] = fmaxf(mx[v], __shfl_xor(mx[v], 8, 16));
  }

  // ---- pass 2: accumulate PV, U1 = (p*St')*V, U3 = p*Vt, l, dot ----
  v8f pv[4] = {}, u1[4] = {}, u3[4] = {};
  float lac[8] = {}, dac[8] = {};
  const int hi = (lane >> 4) * 8;
  const int cc = lane & 15;

  for (int k0 = 0; k0 < SS; k0 += 32) {
#pragma unroll
    for (int j = 0; j < 2; ++j) {
      const int n0 = k0 + j * 16;
      v16bf kb0 = load_bop(Kp  + (size_t)n0 * LDQ, LDQ);
      v16bf kb1 = load_bop(Kp  + (size_t)n0 * LDQ + 32, LDQ);
      v16bf tb0 = load_bop(Ktp + (size_t)n0 * LDQ, LDQ);
      v16bf tb1 = load_bop(Ktp + (size_t)n0 * LDQ + 32, LDQ);
      v8f s = {}, st = {};
      s  = wmma_bf16(qa0, kb0, s);
      s  = wmma_bf16(qa1, kb1, s);
      st = wmma_bf16(qt0, kb0, st);
      st = wmma_bf16(qt1, kb1, st);
      st = wmma_bf16(qa0, tb0, st);
      st = wmma_bf16(qa1, tb1, st);
#pragma unroll
      for (int v = 0; v < 8; ++v) {
        float p = __expf(SCALE * (s[v] - mx[v]));
        float r = p * (SCALE * st[v]);
        lac[v] += p;
        dac[v] += r;
        const int addr = (v + hi) * 32 + j * 16 + cc;
        ldsP[m][addr] = (bf16)p;
        ldsR[m][addr] = (bf16)r;
      }
    }
    // per-wave LDS region, lanes exchange data: enforce store->load ordering
    asm volatile("s_wait_dscnt 0" ::: "memory");
    v16bf pa = load_aop(&ldsP[m][0], 32);
    v16bf ra = load_aop(&ldsR[m][0], 32);
#pragma unroll
    for (int t = 0; t < 4; ++t) {
      v16bf vb = load_bop(Vp + (size_t)t * 16 * SS + k0, SS);
      pv[t] = wmma_bf16(pa, vb, pv[t]);
      u1[t] = wmma_bf16(ra, vb, u1[t]);
      v16bf wb = load_bop(Vtp + (size_t)t * 16 * SS + k0, SS);
      u3[t] = wmma_bf16(pa, wb, u3[t]);
    }
  }

  // ---- reductions and writeback ----
#pragma unroll
  for (int v = 0; v < 8; ++v) {
    lac[v] += __shfl_xor(lac[v], 1, 16);
    lac[v] += __shfl_xor(lac[v], 2, 16);
    lac[v] += __shfl_xor(lac[v], 4, 16);
    lac[v] += __shfl_xor(lac[v], 8, 16);
    dac[v] += __shfl_xor(dac[v], 1, 16);
    dac[v] += __shfl_xor(dac[v], 2, 16);
    dac[v] += __shfl_xor(dac[v], 4, 16);
    dac[v] += __shfl_xor(dac[v], 8, 16);
  }
#pragma unroll
  for (int t = 0; t < 4; ++t)
#pragma unroll
    for (int v = 0; v < 8; ++v) {
      const float linv = 1.0f / lac[v];
      const float d    = dac[v] * linv;
      const int row = q0 + v + hi;
      const int col = h * HD + t * 16 + cc;
      const float ot = linv * (u1[t][v] + u3[t][v] - d * pv[t][v]);
      aot[(((size_t)b * MM + m) * SS + row) * DD + col] = (bf16)ot;
      if (m == 0) ao[((size_t)b * SS + row) * DD + col] = (bf16)(pv[t][v] * linv);
    }
}

// ---------------- host side ----------------
extern "C" void kernel_launch(void* const* d_in, const int* in_sizes, int n_in,
                              void* d_out, int out_size, void* d_ws, size_t ws_size,
                              hipStream_t stream) {
  const float* x     = (const float*)d_in[0];
  const float* xt    = (const float*)d_in[1];
  const float* wqkv  = (const float*)d_in[2];
  const float* wproj = (const float*)d_in[3];
  const float* bproj = (const float*)d_in[4];

  constexpr size_t NX   = (size_t)BB * SS * DD;             // 1,572,864
  constexpr size_t NXT  = (size_t)BB * MM * SS * DD;        // 6,291,456
  constexpr size_t NWQ  = (size_t)3 * DD * DD;              // 1,769,472
  constexpr size_t NWP  = (size_t)DD * DD;                  // 589,824
  constexpr size_t NQKV  = (size_t)BB * SS * LDQ;           // 4,718,592
  constexpr size_t NQKVT = (size_t)BB * MM * SS * LDQ;      // 18,874,368
  constexpr size_t NVT   = (size_t)BB * HH * HD * SS;       // 1,572,864
  constexpr size_t NVTT  = (size_t)BB * MM * HH * HD * SS;  // 6,291,456

  char* w = (char*)d_ws;
  size_t off = 0;
  auto carve = [&](size_t n) {
    bf16* p = (bf16*)(w + off);
    off = (off + n * sizeof(bf16) + 255) & ~(size_t)255;
    return p;
  };
  bf16* xb    = carve(NX);
  bf16* xtb   = carve(NXT);
  bf16* wqkvb = carve(NWQ);
  bf16* wprojb= carve(NWP);
  bf16* qkv   = carve(NQKV);
  bf16* qkvt  = carve(NQKVT);
  bf16* vT    = carve(NVT);
  bf16* vtT   = carve(NVTT);
  bf16* ao    = carve(NX);
  bf16* aot   = carve(NXT);
  (void)ws_size; (void)in_sizes; (void)n_in; (void)out_size;

  // 1) convert inputs to bf16
  k_cvt<<<dim3(2048), dim3(256), 0, stream>>>(x,     xb,    (int)NX);
  k_cvt<<<dim3(4096), dim3(256), 0, stream>>>(xt,    xtb,   (int)NXT);
  k_cvt<<<dim3(2048), dim3(256), 0, stream>>>(wqkv,  wqkvb, (int)NWQ);
  k_cvt<<<dim3(1024), dim3(256), 0, stream>>>(wproj, wprojb,(int)NWP);

  // 2) qkv = x @ Wqkv^T ; qkv_t = x_t @ Wqkv^T
  k_gemm_nt<false><<<dim3(LDQ / 128, (BB * SS) / 128), dim3(256), 0, stream>>>(
      xb, wqkvb, qkv, nullptr, nullptr, LDQ, DD);
  k_gemm_nt<false><<<dim3(LDQ / 128, (BB * MM * SS) / 128), dim3(256), 0, stream>>>(
      xtb, wqkvb, qkvt, nullptr, nullptr, LDQ, DD);

  // 3) transpose V heads: vT[b][h][hd][s], vtT[b][m][h][hd][s]
  k_transpose64<<<dim3(SS / 64, HH, BB), dim3(256), 0, stream>>>(
      qkv + 2 * DD, vT, (size_t)SS * LDQ, (size_t)HH * HD * SS, LDQ, SS);
  k_transpose64<<<dim3(SS / 64, HH, BB * MM), dim3(256), 0, stream>>>(
      qkvt + 2 * DD, vtT, (size_t)SS * LDQ, (size_t)HH * HD * SS, LDQ, SS);

  // 4) fused attention JVP
  k_attn_jvp<<<dim3(SS / 16, HH, BB), dim3(128), 0, stream>>>(qkv, qkvt, vT, vtT, ao, aot);

  // 5) output projection: o = ao @ Wproj^T + b ; o_t = aot @ Wproj^T
  float* o  = (float*)d_out;
  float* ot = o + NX;
  k_gemm_nt<true><<<dim3(DD / 128, (BB * SS) / 128), dim3(256), 0, stream>>>(
      ao, wprojb, nullptr, o, bproj, DD, DD);
  k_gemm_nt<true><<<dim3(DD / 128, (BB * MM * SS) / 128), dim3(256), 0, stream>>>(
      aot, wprojb, nullptr, ot, nullptr, DD, DD);
}
